// NaryTree_67138928771800
// MI455X (gfx1250) — matrix operand order repaired and checked by reading, not compiled, not run on hardware
//
#include <hip/hip_runtime.h>
#include <hip/hip_bf16.h>

// ---------------- problem constants (from reference) ----------------
#define Bn 64
#define Sn 512
#define Tn 12
#define Hn 256
#define En 256
#define KD 256          // contraction dim of every GEMM (E == H == 256)
#define N1 1024         // stage-0 GEMM cols: 768 (iou) + 256 (f)
#define N2 2048         // per-step GEMM cols: 768 r + 768 l + 256 fr + 256 fl
#define KBCT (KD / 32)  // 8 K-blocks of 32

typedef __attribute__((ext_vector_type(16))) __bf16   v16bf;
typedef __attribute__((ext_vector_type(8)))  float    v8f;
typedef __attribute__((ext_vector_type(4)))  unsigned int v4u;

union Frag32 { v4u q[2]; v16bf v; };   // 32 bytes = 16 bf16 per lane

__device__ __forceinline__ float sigmoidf_(float x) { return 1.0f / (1.0f + __expf(-x)); }

// ---------------- utility kernels ----------------
__global__ void k_zero_f32(float* p, int n) {
    int i = blockIdx.x * blockDim.x + threadIdx.x;
    if (i < n) p[i] = 0.0f;
}
__global__ void k_zero_bf16(__bf16* p, int n) {
    int i = blockIdx.x * blockDim.x + threadIdx.x;
    if (i < n) p[i] = (__bf16)0.0f;
}

// x_bf16[b*S+s, :] = bf16(emb[input_ids[b,s], :])
__global__ void k_embed(__bf16* __restrict__ xb, const float* __restrict__ emb,
                        const int* __restrict__ ids) {
    int i = blockIdx.x * blockDim.x + threadIdx.x;
    if (i >= Bn * Sn * En) return;
    int row = i / En, e = i % En;
    xb[i] = (__bf16)emb[(size_t)ids[row] * En + e];
}

// Pre-swizzle weights into WMMA B-fragment layout:
//   packed[((nb*KBCT + kb)*32 + lane)*16 + e]  holds W[k][n] with
//   n = nb*16 + (lane&15),  k = kb*32 + (lane<16 ? 0 : 16) + e
__global__ void k_pack_wx(__bf16* __restrict__ wp, const float* __restrict__ W_ioux,
                          const float* __restrict__ W_f) {
    int idx = blockIdx.x * blockDim.x + threadIdx.x;
    if (idx >= N1 * KD) return;
    int e    = idx & 15;
    int lane = (idx >> 4) & 31;
    int rest = idx >> 9;
    int kb   = rest & (KBCT - 1);
    int nb   = rest / KBCT;
    int n = nb * 16 + (lane & 15);
    int k = kb * 32 + ((lane < 16) ? 0 : 16) + e;
    float v = (n < 768) ? W_ioux[(size_t)k * 768 + n] : W_f[(size_t)k * 256 + (n - 768)];
    wp[idx] = (__bf16)v;
}
__global__ void k_pack_wh(__bf16* __restrict__ wp,
                          const float* __restrict__ U_iou_r, const float* __restrict__ U_iou_l,
                          const float* __restrict__ U_f0, const float* __restrict__ U_f1,
                          const float* __restrict__ U_f2, const float* __restrict__ U_f3) {
    int idx = blockIdx.x * blockDim.x + threadIdx.x;
    if (idx >= N2 * KD) return;
    int e    = idx & 15;
    int lane = (idx >> 4) & 31;
    int rest = idx >> 9;
    int kb   = rest & (KBCT - 1);
    int nb   = rest / KBCT;
    int n = nb * 16 + (lane & 15);
    int k = kb * 32 + ((lane < 16) ? 0 : 16) + e;
    float v;
    if (n < 768)          v = U_iou_r[(size_t)k * 768 + n];
    else if (n < 1536)    v = U_iou_l[(size_t)k * 768 + (n - 768)];
    else if (n < 1792) { int c = n - 1536; v = U_f0[(size_t)k * 256 + c] + U_f1[(size_t)k * 256 + c]; }
    else               { int c = n - 1792; v = U_f2[(size_t)k * 256 + c] + U_f3[(size_t)k * 256 + c]; }
    wp[idx] = (__bf16)v;
}
__global__ void k_bias(float* __restrict__ bias, const float* __restrict__ b_ioux,
                       const float* __restrict__ b_f) {
    int i = blockIdx.x * blockDim.x + threadIdx.x;
    if (i >= N1) return;
    bias[i] = (i < 768) ? b_ioux[i] : b_f[i - 768];
}

// ---------------- bf16 WMMA GEMM ----------------
// C[row, col] = sum_k A[row,k] * W[k,col] (+ bias[col])
// A: row-major bf16 [M x KD]; Wp: pre-swizzled fragment layout; out f32.
// If col < split_col  -> out0[row*split_col + col]
// else                -> out1[row*(N-split_col) + (col-split_col)]
// grid = (M/128, N/64); 256 threads = 8 wave32; wave w owns rows [w*16, w*16+16).
__global__ void __launch_bounds__(256)
k_gemm(const __bf16* __restrict__ A, const __bf16* __restrict__ Wp,
       const float* __restrict__ bias,
       float* __restrict__ out0, float* __restrict__ out1,
       int N, int split_col) {
    const int wave = threadIdx.x >> 5;
    const int lane = threadIdx.x & 31;
    const int m0   = blockIdx.x * 128 + wave * 16;
    const int n0   = blockIdx.y * 64;
    const int hi   = (lane >> 4) & 1;   // which K-half this lane carries
    const int nl   = lane & 15;

    const v8f vz = {0.f, 0.f, 0.f, 0.f, 0.f, 0.f, 0.f, 0.f};
    v8f acc[4];
    acc[0] = vz; acc[1] = vz; acc[2] = vz; acc[3] = vz;

    // A-fragment: lane holds row (m0+nl); K elems {kb*32+hi*16 .. +15}
    const __bf16* aptr = A + (size_t)(m0 + nl) * KD + hi * 16;
    const int nb0 = n0 >> 4;
    const size_t bstride = (size_t)KBCT * 32 * 16;   // stride between N-subtiles

    for (int kb = 0; kb < KBCT; ++kb) {
        Frag32 a;
        a.q[0] = *(const v4u*)(aptr + kb * 32);
        a.q[1] = *(const v4u*)(aptr + kb * 32 + 8);
        if (kb + 1 < KBCT) __builtin_prefetch(aptr + (kb + 1) * 32, 0, 1);
        const __bf16* bbase = Wp + ((size_t)(nb0 * KBCT + kb) * 32 + lane) * 16;
#pragma unroll
        for (int nt = 0; nt < 4; ++nt) {
            Frag32 b;
            const __bf16* bp = bbase + (size_t)nt * bstride;
            b.q[0] = *(const v4u*)(bp);
            b.q[1] = *(const v4u*)(bp + 8);
            acc[nt] = __builtin_amdgcn_wmma_f32_16x16x32_bf16(
                false, a.v, false, b.v, (short)0, acc[nt], false, false);
        }
    }
#pragma unroll
    for (int nt = 0; nt < 4; ++nt) {
        const int col = n0 + nt * 16 + nl;
        const float bv = bias ? bias[col] : 0.0f;
#pragma unroll
        for (int r = 0; r < 8; ++r) {
            const int row = m0 + hi * 8 + r;   // C/D layout: VGPR r -> M=r (+8 for hi lanes)
            const float v = acc[nt][r] + bv;
            if (col < split_col) out0[(size_t)row * split_col + col] = v;
            else                 out1[(size_t)row * (N - split_col) + (col - split_col)] = v;
        }
    }
}

// ---------------- per-step scatter/gather/pointwise ----------------
__global__ void k_count(float* __restrict__ cnt, const int* __restrict__ td_ids, int t) {
    int i = blockIdx.x * blockDim.x + threadIdx.x;
    if (i >= Bn * Sn) return;
    int b = i / Sn, j = i % Sn;
    int td = td_ids[((size_t)b * Tn + t) * Sn + j];
    atomicAdd(&cnt[b * Sn + td], 1.0f);
}

// iou_ws[b, tr, :768] += P[node, 0:768]; iou_ws[b, tl, :768] += P[node, 768:1536]
__global__ void k_scatter_iou(float* __restrict__ iou, const float* __restrict__ P,
                              const int* __restrict__ tr_ids, const int* __restrict__ tl_ids, int t) {
    long long i = (long long)blockIdx.x * blockDim.x + threadIdx.x;
    if (i >= (long long)Bn * Sn * 768) return;
    int node = (int)(i / 768);
    int col  = (int)(i % 768);
    int b = node / Sn, j = node % Sn;
    size_t idb = ((size_t)b * Tn + t) * Sn + j;
    int tr = tr_ids[idb], tl = tl_ids[idb];
    const float* prow = P + (size_t)node * N2;
    atomicAdd(&iou[((size_t)b * Sn + tr) * 768 + col], prow[col]);
    atomicAdd(&iou[((size_t)b * Sn + tl) * 768 + col], prow[768 + col]);
}

// cnew = sigmoid(i) * tanh(u)
__global__ void k_iu(float* __restrict__ cnew, const float* __restrict__ iou) {
    int i = blockIdx.x * blockDim.x + threadIdx.x;
    if (i >= Bn * Sn * Hn) return;
    int node = i / Hn, col = i % Hn;
    const float* r = iou + (size_t)node * 768;
    cnew[i] = sigmoidf_(r[col]) * tanhf(r[512 + col]);
}

// f = sigmoid(f_x[td] + pfr[tr] + pfl[tl]); cnew[td] += f * c_old[child]
__global__ void k_fc(float* __restrict__ cnew, const float* __restrict__ f_x,
                     const float* __restrict__ P, const float* __restrict__ c,
                     const int* __restrict__ td_ids, const int* __restrict__ tr_ids,
                     const int* __restrict__ tl_ids, int t) {
    int i = blockIdx.x * blockDim.x + threadIdx.x;
    if (i >= Bn * Sn * Hn) return;
    int node = i / Hn, col = i % Hn;
    int b = node / Sn, j = node % Sn;
    size_t idb = ((size_t)b * Tn + t) * Sn + j;
    int td = td_ids[idb], tr = tr_ids[idb], tl = tl_ids[idb];
    float f = sigmoidf_(f_x[((size_t)b * Sn + td) * Hn + col]
                      + P[((size_t)b * Sn + tr) * N2 + 1536 + col]
                      + P[((size_t)b * Sn + tl) * N2 + 1792 + col]);
    atomicAdd(&cnew[((size_t)b * Sn + td) * Hn + col], f * c[i]);
}

// masked update: h = o * tanh(c_new); keep bf16 mirror of h for next GEMM
__global__ void k_update(float* __restrict__ h, float* __restrict__ c, __bf16* __restrict__ hb,
                         const float* __restrict__ cnew, const float* __restrict__ iou,
                         const float* __restrict__ cnt) {
    int i = blockIdx.x * blockDim.x + threadIdx.x;
    if (i >= Bn * Sn * Hn) return;
    int node = i / Hn, col = i % Hn;
    int s = node % Sn;
    if (s == 0 || cnt[node] <= 0.0f) return;   // mask: node 0 excluded; only parents with kids
    float cn = cnew[i];
    float o  = sigmoidf_(iou[(size_t)node * 768 + 256 + col]);
    float hn = o * tanhf(cn);
    c[i] = cn;
    h[i] = hn;
    hb[i] = (__bf16)hn;
}

// root = max_j tree_ids[b, T-1, j]; h_root[b,:] = h[b, root, :]  (LDS max-reduce)
__global__ void __launch_bounds__(256)
k_root(float* __restrict__ hroot, const float* __restrict__ h, const int* __restrict__ td_ids) {
    __shared__ int sm[256];
    int b = blockIdx.x, t = threadIdx.x;
    const int* row = td_ids + ((size_t)b * Tn + (Tn - 1)) * Sn;
    int m  = row[t];
    int m2 = row[t + 256];
    if (m2 > m) m = m2;
    sm[t] = m;
    __syncthreads();
    for (int s = 128; s > 0; s >>= 1) {
        if (t < s && sm[t + s] > sm[t]) sm[t] = sm[t + s];
        __syncthreads();
    }
    int root = sm[0];
    hroot[b * Hn + t] = h[((size_t)b * Sn + root) * Hn + t];
}

// ---------------- host driver ----------------
#define CDIV(a, b) (((a) + (b) - 1) / (b))

extern "C" void kernel_launch(void* const* d_in, const int* in_sizes, int n_in,
                              void* d_out, int out_size, void* d_ws, size_t ws_size,
                              hipStream_t stream) {
    const int*   input_ids = (const int*)d_in[0];
    const int*   tree_ids  = (const int*)d_in[1];
    const int*   tree_r    = (const int*)d_in[2];
    const int*   tree_l    = (const int*)d_in[3];
    const float* emb       = (const float*)d_in[4];
    const float* W_ioux    = (const float*)d_in[5];
    const float* b_ioux    = (const float*)d_in[6];
    const float* U_iou_r   = (const float*)d_in[7];
    const float* U_iou_l   = (const float*)d_in[8];
    const float* W_f       = (const float*)d_in[9];
    const float* b_f       = (const float*)d_in[10];
    const float* U_f0      = (const float*)d_in[11];
    const float* U_f1      = (const float*)d_in[12];
    const float* U_f2      = (const float*)d_in[13];
    const float* U_f3      = (const float*)d_in[14];

    const int M = Bn * Sn;                 // 32768 rows in every GEMM
    float* h_out  = (float*)d_out;                         // [B,S,H]
    float* c_out  = h_out + (size_t)M * Hn;                // [B,S,H]
    float* hr_out = c_out + (size_t)M * Hn;                // [B,H]

    // workspace carving
    char* ws = (char*)d_ws;
    size_t off = 0;
    auto carve = [&](size_t bytes) -> char* {
        char* p = ws + off;
        off += (bytes + 255) & ~(size_t)255;
        return p;
    };
    __bf16* x_bf   = (__bf16*)carve((size_t)M * En * 2);
    __bf16* h_bf   = (__bf16*)carve((size_t)M * Hn * 2);
    __bf16* wx_pk  = (__bf16*)carve((size_t)KD * N1 * 2);
    __bf16* wh_pk  = (__bf16*)carve((size_t)KD * N2 * 2);
    float*  bias_x = (float*)carve((size_t)N1 * 4);
    float*  iou_x  = (float*)carve((size_t)M * 768 * 4);
    float*  f_x    = (float*)carve((size_t)M * Hn * 4);
    float*  P      = (float*)carve((size_t)M * N2 * 4);
    float*  iou_ws = (float*)carve((size_t)M * 768 * 4);
    float*  cnew   = (float*)carve((size_t)M * Hn * 4);
    float*  cnt    = (float*)carve((size_t)M * 4);

    const int TPB = 256;
    const int nBSH = M * Hn;                 // 8,388,608

    // init h = c = 0 (d_out is poisoned), h_bf16 = 0
    k_zero_f32<<<CDIV(nBSH, TPB), TPB, 0, stream>>>(h_out, nBSH);
    k_zero_f32<<<CDIV(nBSH, TPB), TPB, 0, stream>>>(c_out, nBSH);
    k_zero_bf16<<<CDIV(nBSH, TPB), TPB, 0, stream>>>(h_bf, nBSH);

    // stage 0: embed + pack weights + bias + GEMM1 (iou_x | f_x)
    k_embed<<<CDIV(M * En, TPB), TPB, 0, stream>>>(x_bf, emb, input_ids);
    k_pack_wx<<<CDIV(N1 * KD, TPB), TPB, 0, stream>>>(wx_pk, W_ioux, W_f);
    k_pack_wh<<<CDIV(N2 * KD, TPB), TPB, 0, stream>>>(wh_pk, U_iou_r, U_iou_l, U_f0, U_f1, U_f2, U_f3);
    k_bias<<<CDIV(N1, TPB), TPB, 0, stream>>>(bias_x, b_ioux, b_f);
    k_gemm<<<dim3(M / 128, N1 / 64), TPB, 0, stream>>>(x_bf, wx_pk, bias_x, iou_x, f_x, N1, 768);

    // T scan steps
    for (int t = 0; t < Tn; ++t) {
        // P = h @ [U_iou_r | U_iou_l | U_f0+U_f1 | U_f2+U_f3]
        k_gemm<<<dim3(M / 128, N2 / 64), TPB, 0, stream>>>(h_bf, wh_pk, nullptr, P, P, N2, N2);
        // iou_ws = iou_x (fresh copy each step)
        hipMemcpyAsync(iou_ws, iou_x, (size_t)M * 768 * 4, hipMemcpyDeviceToDevice, stream);
        k_zero_f32<<<CDIV(M, TPB), TPB, 0, stream>>>(cnt, M);
        k_count<<<CDIV(M, TPB), TPB, 0, stream>>>(cnt, tree_ids, t);
        k_scatter_iou<<<CDIV(M * 768, TPB), TPB, 0, stream>>>(iou_ws, P, tree_r, tree_l, t);
        k_iu<<<CDIV(nBSH, TPB), TPB, 0, stream>>>(cnew, iou_ws);
        k_fc<<<CDIV(nBSH, TPB), TPB, 0, stream>>>(cnew, f_x, P, c_out, tree_ids, tree_r, tree_l, t);
        k_update<<<CDIV(nBSH, TPB), TPB, 0, stream>>>(h_out, c_out, h_bf, cnew, iou_ws, cnt);
    }

    // h_root
    k_root<<<Bn, 256, 0, stream>>>(hr_out, h_out, tree_ids);
}
